// LocalTransformer_28389733827159
// MI455X (gfx1250) — matrix operand order repaired and checked by reading, not compiled, not run on hardware
//
#include <hip/hip_runtime.h>
#include <hip/hip_fp16.h>

// ---------------------------------------------------------------------------
// Types / WMMA helpers (CDNA5 gfx1250, wave32)
// ---------------------------------------------------------------------------
typedef _Float16 h16;
typedef __attribute__((ext_vector_type(16))) _Float16 v16h;
typedef __attribute__((ext_vector_type(8)))  float    v8f;

#define WMMA(a, b, c) __builtin_amdgcn_wmma_f32_16x16x32_f16( \
    false, (a), false, (b), (short)0, (c), false, false)

// ---- CDNA5 async memory->LDS copy (ISA 10.x / 15.18.3, ASYNCcnt) ----------
// GV mode: per-lane 16B global -> per-lane LDS address, no VGPR round-trip.
// INST_OFFSET is added to BOTH the LDS and global addresses (ISA 4.4), so
// consecutive 16B chunks share one address setup via the offset: immediate.
// LDS address = low 32 bits of the generic pointer (flat->LDS: addr[31:0]).
#define ASYNC_B128_OFF(lds, g, OFF)                                        \
    asm volatile("global_load_async_to_lds_b128 %0, %1, off offset:" #OFF \
                 :: "v"((unsigned)(size_t)(const void*)(lds)),            \
                    "v"((const void*)(g)) : "memory")

#define ASYNC_B128_X4(lds, g)          \
    do {                               \
        ASYNC_B128_OFF((lds), (g), 0); \
        ASYNC_B128_OFF((lds), (g), 16);\
        ASYNC_B128_OFF((lds), (g), 32);\
        ASYNC_B128_OFF((lds), (g), 48);\
    } while (0)

__device__ __forceinline__ void wait_async_lds() {
    asm volatile("s_wait_asynccnt 0x0" ::: "memory");
}

// A fragment: 16x32 f16 tile, row-major source with given stride (in halves).
// Layout (ISA 7.12.2): lane L<16 holds row L, K=0..7 (elems 0-7) and K=16..23
// (elems 8-15); lane 16+L holds row L, K=8..15 and K=24..31.
__device__ __forceinline__ v16h load_a_frag(const h16* p, int stride, int lane) {
    const int row = lane & 15;
    const int kh  = (lane >> 4) * 8;
    v16h f;
#pragma unroll
    for (int i = 0; i < 8; ++i) f[i]     = p[row * stride + kh + i];
#pragma unroll
    for (int i = 0; i < 8; ++i) f[i + 8] = p[row * stride + 16 + kh + i];
    return f;
}

// B fragment from a TRANSPOSED tile M[n][k] (row-major, stride in halves):
// B[k][n] = M[n][k]. Lanes 0-15 hold column n=lane with K=0..7 and K=16..23,
// lanes 16-31 hold K=8..15 and K=24..31 -> 2 runs of 8 halves -> 2x ds_load_b128.
__device__ __forceinline__ v16h load_bT_frag(const h16* p, int stride, int lane) {
    const int col = lane & 15;          // n
    const int kb  = (lane >> 4) * 8;    // k base
    v16h f;
#pragma unroll
    for (int i = 0; i < 8; ++i) { f[i] = p[col * stride + kb + i]; f[i + 8] = p[col * stride + 16 + kb + i]; }
    return f;
}

// ---------------------------------------------------------------------------
// Problem constants
// ---------------------------------------------------------------------------
#define NTOK   8192
#define DIMM   512
#define HEADS  8
#define DH     64
#define INNER  512
#define QKVW   1536
#define WS     256
#define NWIN   32
#define FFN    1365
#define FFNP   1408   /* FFN padded to a multiple of 64 (GEMM K-step) */
#define HLD    1408   /* f32 row stride for Ha/Hg scratch */
#define DEPTH  6

// ---------------------------------------------------------------------------
// Generic f16 GEMM: C[M,N](f32) = A[M,K](f16) x B[K,N](f16) (+bias)(+residual)
// B is supplied TRANSPOSED: BT[N][K] row-major with leading dim ldbt.
// REQUIRES: K % 64 == 0, M % 128 == 0, lda/ldbt rows cover K, 16B alignment.
// Block tile 128x128, K-step 64, 256 threads = 8 waves (4 in M x 2 in N).
// Double-buffered LDS staging: async global->LDS b128 copies for tile ks+1
// are in flight while tile ks feeds 16 WMMAs; one s_wait_asynccnt + barrier
// per K-step.
// ---------------------------------------------------------------------------
__global__ __launch_bounds__(256) void gemm_f16f32(
    const h16* __restrict__ A, int lda,
    const h16* __restrict__ BT, int ldbt,
    float* __restrict__ C, int ldc,
    const float* __restrict__ bias,
    const float* __restrict__ residual,
    int M, int N, int K)
{
    __shared__ h16 sA[2][128 * 64];
    __shared__ h16 sBT[2][128 * 64];

    const int tid  = threadIdx.x;
    const int lane = tid & 31;
    const int wid  = tid >> 5;
    const int wm   = wid & 3;   // wave row (32 rows each)
    const int wn   = wid >> 2;  // wave col (64 cols each)
    const int bm   = blockIdx.x * 128;
    const int bn   = blockIdx.y * 128;

    // Staging coordinates (each thread moves 32 contiguous halves = 4x b128)
    const int srow = tid >> 1;
    const int sseg = (tid & 1) * 32;
    const bool nok = (bn + srow) < N;

    const h16* srcA0 = A + (size_t)(bm + srow) * lda + sseg;
    const h16* srcB0 = BT + (size_t)(bn + srow) * ldbt + sseg;

    auto stage = [&](int ks, int buf) {
        const int k0 = ks * 64;
        {
            h16* dst = &sA[buf][srow * 64 + sseg];
            ASYNC_B128_X4(dst, srcA0 + k0);
        }
        {
            h16* dst = &sBT[buf][srow * 64 + sseg];
            if (nok) {
                ASYNC_B128_X4(dst, srcB0 + k0);
            } else {
                const uint4 z = {0u, 0u, 0u, 0u};
#pragma unroll
                for (int i = 0; i < 4; ++i)
                    reinterpret_cast<uint4*>(dst)[i] = z;
            }
        }
    };

    v8f acc[2][4];
#pragma unroll
    for (int i = 0; i < 2; ++i)
#pragma unroll
        for (int j = 0; j < 4; ++j) { v8f z = {}; acc[i][j] = z; }

    const int ksteps = K >> 6;
    stage(0, 0);
    wait_async_lds();
    __syncthreads();

    for (int ks = 0; ks < ksteps; ++ks) {
        const int buf = ks & 1;
        if (ks + 1 < ksteps)              // async-stage next tile into other buf
            stage(ks + 1, buf ^ 1);

#pragma unroll
        for (int ksl = 0; ksl < 2; ++ksl) {
            v16h af[2], bf[4];
#pragma unroll
            for (int mt = 0; mt < 2; ++mt)
                af[mt] = load_a_frag(&sA[buf][(wm * 32 + mt * 16) * 64 + ksl * 32], 64, lane);
#pragma unroll
            for (int nt = 0; nt < 4; ++nt)
                bf[nt] = load_bT_frag(&sBT[buf][(wn * 64 + nt * 16) * 64 + ksl * 32], 64, lane);
#pragma unroll
            for (int mt = 0; mt < 2; ++mt)
#pragma unroll
                for (int nt = 0; nt < 4; ++nt)
                    acc[mt][nt] = WMMA(af[mt], bf[nt], acc[mt][nt]);
        }
        wait_async_lds();   // next tile arrived (overlapped with WMMAs above)
        __syncthreads();
    }

    // Epilogue: C-layout row = r + (lane>>4)*8, col = lane&15
    const int rbase = (lane >> 4) * 8;
    const int cl    = lane & 15;
#pragma unroll
    for (int mt = 0; mt < 2; ++mt) {
#pragma unroll
        for (int nt = 0; nt < 4; ++nt) {
            const int col = bn + wn * 64 + nt * 16 + cl;
            if (col >= N) continue;
            const float badd = bias ? bias[col] : 0.f;
#pragma unroll
            for (int r = 0; r < 8; ++r) {
                const int row = bm + wm * 32 + mt * 16 + rbase + r;
                float v = acc[mt][nt][r] + badd;
                if (residual) v += residual[(size_t)row * ldc + col];
                C[(size_t)row * ldc + col] = v;
            }
        }
    }
}

// ---------------------------------------------------------------------------
// QKV prep: per (token n, head h) — L2-normalize q/k over dh, apply learned
// per-dim scales, apply xPos rotary; emit q/k as f16 in [h][n][dh] layout,
// and V TRANSPOSED [h][dh][n] so the attention kernel stages it contiguously.
// ---------------------------------------------------------------------------
__global__ __launch_bounds__(256) void qkv_prep(
    const float* __restrict__ qkv,           // (8192, 1536)
    const float* __restrict__ qs,            // (64)
    const float* __restrict__ kscl,          // (64)
    h16* __restrict__ qh, h16* __restrict__ kh, h16* __restrict__ vt)
{
    const int idx = blockIdx.x * blockDim.x + threadIdx.x; // 0..65535
    const int hh  = idx & 7;
    const int n   = idx >> 3;
    const float* base = qkv + (size_t)n * QKVW + hh * DH;

    float sq = 0.f, sk = 0.f;
#pragma unroll 8
    for (int d = 0; d < DH; ++d) {
        const float a = base[d];         sq += a * a;
        const float b = base[INNER + d]; sk += b * b;
    }
    const float rq = 1.f / fmaxf(sqrtf(sq), 1e-12f);
    const float rk = 1.f / fmaxf(sqrtf(sk), 1e-12f);

    const float t     = (float)n;
    const float power = (t - (float)(NTOK / 2)) / (float)(WS / 2);
    const size_t ob   = (size_t)hh * NTOK * DH + (size_t)n * DH;

#pragma unroll 4
    for (int j = 0; j < DH / 2; ++j) {
        const float invf = __powf(10000.f, -(2.f * j) / (float)DH);
        const float ang  = t * invf;
        const float c = __cosf(ang), s = __sinf(ang);
        const float sv  = (2.f * j + 0.4f * DH) / (1.4f * DH);
        const float xs  = __powf(sv, power);
        const float xsi = 1.f / xs;

        const float q1 = base[j]            * rq * qs[j];
        const float q2 = base[j + DH/2]     * rq * qs[j + DH/2];
        const float k1 = base[INNER + j]        * rk * kscl[j];
        const float k2 = base[INNER + j + DH/2] * rk * kscl[j + DH/2];

        qh[ob + j]        = (h16)((q1 * c - q2 * s) * xs);
        qh[ob + j + DH/2] = (h16)((q2 * c + q1 * s) * xs);
        kh[ob + j]        = (h16)((k1 * c - k2 * s) * xsi);
        kh[ob + j + DH/2] = (h16)((k2 * c + k1 * s) * xsi);
    }
    // V transposed: vt[h][dh][n]
    const size_t vb = (size_t)hh * DH * NTOK + n;
#pragma unroll 8
    for (int d = 0; d < DH; ++d)
        vt[vb + (size_t)d * NTOK] = (h16)base[2 * INNER + d];
}

// ---------------------------------------------------------------------------
// Local attention, flash-style. Block = one (window w, head h); 8 waves.
// Wave owns 32 query rows (2 M-tiles of 16, outer loop). Keys: 512 in 16
// chunks of 32 staged in LDS via async b128 copies (K as [key][dh], V as
// [dh][key] so both WMMA B fragments are contiguous per lane).
// ---------------------------------------------------------------------------
__global__ __launch_bounds__(256) void local_attn(
    const h16* __restrict__ qh, const h16* __restrict__ kh,
    const h16* __restrict__ vt, h16* __restrict__ out /* (8192, 512) */)
{
    const int w  = blockIdx.x;  // 0..31
    const int h  = blockIdx.y;  // 0..7
    __shared__ h16 sK[32 * 64];   // [key][dh]
    __shared__ h16 sVT[64 * 32];  // [dh][key]
    __shared__ h16 sP[8][16 * 32];

    const int tid = threadIdx.x, lane = tid & 31, wid = tid >> 5;
    const size_t hb  = (size_t)h * NTOK * DH;
    const h16* qhh = qh + hb;
    const h16* khh = kh + hb;
    const h16* vth = vt + hb;   // [dh][n]

    const int rbase = (lane >> 4) * 8;
    const int cl    = lane & 15;

    for (int mt = 0; mt < 2; ++mt) {
        const int qrow0 = w * WS + wid * 32 + mt * 16;
        v16h qf[2];
#pragma unroll
        for (int ksl = 0; ksl < 2; ++ksl)
            qf[ksl] = load_a_frag(qhh + (size_t)qrow0 * DH + ksl * 32, DH, lane);

        v8f acc[4];
#pragma unroll
        for (int nt = 0; nt < 4; ++nt) { v8f z = {}; acc[nt] = z; }
        float mrow[8], lrow[8];
#pragma unroll
        for (int r = 0; r < 8; ++r) { mrow[r] = -1e30f; lrow[r] = 0.f; }

        for (int chunk = 0; chunk < 16; ++chunk) {
            const int kstart = (w - 1) * WS + chunk * 32; // global key of chunk
            // Stage K [key][dh]: one async b128 per thread, uniform guard
            {
                const int key = tid >> 3;
                const int d0  = (tid & 7) * 8;
                const int gk  = kstart + key;
                h16* dst = sK + key * 64 + d0;
                if (gk >= 0) {
                    ASYNC_B128_OFF(dst, khh + (size_t)gk * DH + d0, 0);
                } else {
                    const uint4 z = {0u, 0u, 0u, 0u};
                    *reinterpret_cast<uint4*>(dst) = z;
                }
            }
            // Stage V^T [dh][key]: one async b128 per thread, uniform guard
            // (kstart is a multiple of 32, runs are 8-aligned -> sign uniform)
            {
                const int d  = tid >> 2;        // 0..63
                const int s8 = (tid & 3) * 8;   // 0,8,16,24
                h16* dst = sVT + d * 32 + s8;
                if (kstart + s8 >= 0) {
                    ASYNC_B128_OFF(dst, vth + (size_t)d * NTOK + kstart + s8, 0);
                } else {
                    const uint4 z = {0u, 0u, 0u, 0u};
                    *reinterpret_cast<uint4*>(dst) = z;
                }
            }
            wait_async_lds();
            __syncthreads();

            // S = q . k^T  (two 16x16 score tiles: keys 0-15 and 16-31)
            v8f s0 = {}, s1 = {};
#pragma unroll
            for (int ksl = 0; ksl < 2; ++ksl) {
                v16h b0 = load_bT_frag(sK + 0 * 64  + ksl * 32, 64, lane);
                v16h b1 = load_bT_frag(sK + 16 * 64 + ksl * 32, 64, lane);
                s0 = WMMA(qf[ksl], b0, s0);
                s1 = WMMA(qf[ksl], b1, s1);
            }

            // scale + mask  (causal, window<=256, pad tk>=0)
            const int tk0 = kstart + cl;
            const int tk1 = kstart + 16 + cl;
            float p0[8], p1[8], rm[8];
#pragma unroll
            for (int r = 0; r < 8; ++r) {
                const int tq = qrow0 + rbase + r;
                float v0 = s0[r] * 8.f, v1 = s1[r] * 8.f;
                const int d0 = tq - tk0, d1 = tq - tk1;
                if (!(d0 >= 0 && d0 <= WS && tk0 >= 0)) v0 = -1e30f;
                if (!(d1 >= 0 && d1 <= WS && tk1 >= 0)) v1 = -1e30f;
                p0[r] = v0; p1[r] = v1;
                rm[r] = fmaxf(v0, v1);
            }
            // row reductions across the 16 lanes sharing each row
#pragma unroll
            for (int x = 1; x < 16; x <<= 1)
#pragma unroll
                for (int r = 0; r < 8; ++r)
                    rm[r] = fmaxf(rm[r], __shfl_xor(rm[r], x, 32));

            float alpha[8], rs[8];
#pragma unroll
            for (int r = 0; r < 8; ++r) {
                const float mnew = fmaxf(mrow[r], rm[r]);
                alpha[r] = __expf(mrow[r] - mnew);
                mrow[r]  = mnew;
                p0[r] = __expf(p0[r] - mnew);
                p1[r] = __expf(p1[r] - mnew);
                rs[r] = p0[r] + p1[r];
            }
#pragma unroll
            for (int x = 1; x < 16; x <<= 1)
#pragma unroll
                for (int r = 0; r < 8; ++r)
                    rs[r] += __shfl_xor(rs[r], x, 32);
#pragma unroll
            for (int r = 0; r < 8; ++r)
                lrow[r] = lrow[r] * alpha[r] + rs[r];

            // write P (C-layout -> row-major 16x32) and rescale accumulators
            h16* Pw = sP[wid];
#pragma unroll
            for (int r = 0; r < 8; ++r) {
                const int row = rbase + r;
                Pw[row * 32 + cl]      = (h16)p0[r];
                Pw[row * 32 + 16 + cl] = (h16)p1[r];
            }
#pragma unroll
            for (int nt = 0; nt < 4; ++nt)
#pragma unroll
                for (int r = 0; r < 8; ++r)
                    acc[nt][r] *= alpha[r];
            __syncthreads();

            // O += P x V   (A = P 16x32, B from sVT -> contiguous per lane)
            v16h ap = load_a_frag(Pw, 32, lane);
#pragma unroll
            for (int nt = 0; nt < 4; ++nt) {
                v16h bv = load_bT_frag(sVT + (nt * 16) * 32, 32, lane);
                acc[nt] = WMMA(ap, bv, acc[nt]);
            }
            __syncthreads();
        }

        // normalize and emit f16 into (n, h*64+dh) layout for the Wo GEMM
#pragma unroll
        for (int nt = 0; nt < 4; ++nt)
#pragma unroll
            for (int r = 0; r < 8; ++r) {
                const int n  = qrow0 + rbase + r;
                const int dc = nt * 16 + cl;
                out[(size_t)n * INNER + h * DH + dc] =
                    (h16)(acc[nt][r] / fmaxf(lrow[r], 1e-20f));
            }
    }
}

// ---------------------------------------------------------------------------
// Elementwise kernels
// ---------------------------------------------------------------------------
__global__ void f32_to_f16(const float* __restrict__ s, h16* __restrict__ d, int n) {
    int i = blockIdx.x * blockDim.x + threadIdx.x;
    if (i < n) d[i] = (h16)s[i];
}
__global__ void copy_f32(const float* __restrict__ s, float* __restrict__ d, int n) {
    int i = blockIdx.x * blockDim.x + threadIdx.x;
    if (i < n) d[i] = s[i];
}
// Transposing converter: src (L, K, N) f32 -> dst (L, N, K) f16
__global__ void convT(const float* __restrict__ s, h16* __restrict__ d,
                      int K, int N, int total) {
    int i = blockIdx.x * blockDim.x + threadIdx.x;
    if (i >= total) return;
    const int k = i % K;
    const int n = (i / K) % N;
    const int l = i / (K * N);
    d[i] = (h16)s[((size_t)l * K + k) * N + n];
}
// W2: (L,1365,512) f32 -> transposed+padded (L,512,1408) f16
__global__ void convT_w2(const float* __restrict__ s, h16* __restrict__ d, int total) {
    int i = blockIdx.x * blockDim.x + threadIdx.x;
    if (i >= total) return;
    const int k = i % FFNP;
    const int n = (i / FFNP) % DIMM;
    const int l = i / (FFNP * DIMM);
    d[i] = (k < FFN) ? (h16)s[((size_t)l * FFN + k) * DIMM + n] : (h16)0.f;
}
// out[n, 0..1407] = a * gelu_exact(g), zero-padded cols 1365..1407
__global__ void gelu_gate(const float* __restrict__ Ha, const float* __restrict__ Hg,
                          h16* __restrict__ o, int total) {
    int i = blockIdx.x * blockDim.x + threadIdx.x;
    if (i >= total) return;
    const int col = i % FFNP;
    const int row = i / FFNP;
    float v = 0.f;
    if (col < FFN) {
        const float a = Ha[(size_t)row * HLD + col];
        const float g = Hg[(size_t)row * HLD + col];
        v = a * (0.5f * g * (1.f + erff(g * 0.70710678118f)));
    }
    o[i] = (h16)v;
}

// ---------------------------------------------------------------------------
// Launch
// ---------------------------------------------------------------------------
static inline size_t align_up(size_t x) { return (x + 255) & ~(size_t)255; }

extern "C" void kernel_launch(void* const* d_in, const int* in_sizes, int n_in,
                              void* d_out, int out_size, void* d_ws, size_t ws_size,
                              hipStream_t stream) {
    (void)in_sizes; (void)n_in; (void)ws_size;
    const float* x      = (const float*)d_in[0];
    const float* Wqkv   = (const float*)d_in[1];
    const float* Wo     = (const float*)d_in[2];
    const float* qscale = (const float*)d_in[3];
    const float* kscale = (const float*)d_in[4];
    const float* W1     = (const float*)d_in[5];
    const float* b1     = (const float*)d_in[6];
    const float* W2     = (const float*)d_in[7];
    const float* b2     = (const float*)d_in[8];

    char* ws = (char*)d_ws;
    size_t off = 0;
    auto carve = [&](size_t bytes) { void* p = ws + off; off += align_up(bytes); return p; };

    float* resid   = (float*)carve((size_t)NTOK * DIMM * 4);
    h16*   act16   = (h16*)  carve((size_t)NTOK * DIMM * 2);
    float* qkvF    = (float*)carve((size_t)NTOK * QKVW * 4);   // aliased as Ha
    float* HgF     = (float*)carve((size_t)NTOK * HLD * 4);
    h16*   hg16    = (h16*)  carve((size_t)NTOK * FFNP * 2);
    h16*   qh16    = (h16*)  carve((size_t)HEADS * NTOK * DH * 2);
    h16*   kh16    = (h16*)  carve((size_t)HEADS * NTOK * DH * 2);
    h16*   vt16    = (h16*)  carve((size_t)HEADS * DH * NTOK * 2);
    h16*   attn16  = (h16*)  carve((size_t)NTOK * INNER * 2);
    h16*   WqkvT   = (h16*)  carve((size_t)DEPTH * QKVW * DIMM * 2);
    h16*   WoT     = (h16*)  carve((size_t)DEPTH * DIMM * INNER * 2);
    h16*   W1T     = (h16*)  carve((size_t)DEPTH * 2 * FFN * DIMM * 2);
    h16*   W2T     = (h16*)  carve((size_t)DEPTH * DIMM * FFNP * 2);
    float* HaF     = qkvF;   // reuse: qkv dead once heads are prepped

    const int T = 256;
    auto blks = [](int n, int t) { return (n + t - 1) / t; };

    // One-time (per launch) weight conversion to transposed f16
    { int n = DEPTH * QKVW * DIMM;  convT<<<blks(n,T),T,0,stream>>>(Wqkv, WqkvT, DIMM, QKVW, n); }
    { int n = DEPTH * DIMM * INNER; convT<<<blks(n,T),T,0,stream>>>(Wo,   WoT,   INNER, DIMM, n); }
    { int n = DEPTH * 2*FFN * DIMM; convT<<<blks(n,T),T,0,stream>>>(W1,   W1T,   DIMM, 2*FFN, n); }
    { int n = DEPTH * DIMM * FFNP;  convT_w2<<<blks(n,T),T,0,stream>>>(W2, W2T, n); }
    { int n = NTOK * DIMM;          copy_f32<<<blks(n,T),T,0,stream>>>(x, resid, n); }

    for (int i = 0; i < DEPTH; ++i) {
        // ---- attention block ----
        { int n = NTOK * DIMM; f32_to_f16<<<blks(n,T),T,0,stream>>>(resid, act16, n); }
        gemm_f16f32<<<dim3(NTOK/128, QKVW/128), T, 0, stream>>>(
            act16, DIMM, WqkvT + (size_t)i * QKVW * DIMM, DIMM,
            qkvF, QKVW, nullptr, nullptr, NTOK, QKVW, DIMM);
        qkv_prep<<<blks(NTOK * HEADS, T), T, 0, stream>>>(
            qkvF, qscale + i * DH, kscale + i * DH, qh16, kh16, vt16);
        local_attn<<<dim3(NWIN, HEADS), T, 0, stream>>>(qh16, kh16, vt16, attn16);
        gemm_f16f32<<<dim3(NTOK/128, DIMM/128), T, 0, stream>>>(
            attn16, INNER, WoT + (size_t)i * DIMM * INNER, INNER,
            resid, DIMM, nullptr, resid, NTOK, DIMM, INNER);

        // ---- feedforward block (SwiGLU-style) ----
        { int n = NTOK * DIMM; f32_to_f16<<<blks(n,T),T,0,stream>>>(resid, act16, n); }
        gemm_f16f32<<<dim3(NTOK/128, blks(FFN,128)), T, 0, stream>>>(
            act16, DIMM, W1T + (size_t)i * 2 * FFN * DIMM, DIMM,
            HaF, HLD, b1 + (size_t)i * 2 * FFN, nullptr, NTOK, FFN, DIMM);
        gemm_f16f32<<<dim3(NTOK/128, blks(FFN,128)), T, 0, stream>>>(
            act16, DIMM, W1T + (size_t)i * 2 * FFN * DIMM + (size_t)FFN * DIMM, DIMM,
            HgF, HLD, b1 + (size_t)i * 2 * FFN + FFN, nullptr, NTOK, FFN, DIMM);
        { int n = NTOK * FFNP; gelu_gate<<<blks(n,T),T,0,stream>>>(HaF, HgF, hg16, n); }
        gemm_f16f32<<<dim3(NTOK/128, DIMM/128), T, 0, stream>>>(
            hg16, FFNP, W2T + (size_t)i * DIMM * FFNP, FFNP,
            resid, DIMM, b2 + (size_t)i * DIMM, resid, NTOK, DIMM, FFNP);
    }

    copy_f32<<<blks(out_size, T), T, 0, stream>>>(resid, (float*)d_out, out_size);
}
